// MultiQuerySelfAttention_78426102825680
// MI455X (gfx1250) — matrix operand order repaired and checked by reading, not compiled, not run on hardware
//
#include <hip/hip_runtime.h>
#include <hip/hip_bf16.h>

// MQA: x(4,2048,1024) -> Q/K/V proj -> 1-KV-head attention -> O proj.
// All GEMMs in bf16 via v_wmma_f32_16x16x32_bf16, fp32 accumulate.
// Staging uses gfx1250 async global->LDS DMA (ASYNCcnt), double-buffered so
// the DMA engine runs concurrently with the WMMA pipeline.

#define D_MODEL 1024
#define NHQ     16
#define DH      64
#define BATCH   4
#define SEQ     2048
#define ROWS    (BATCH * SEQ)   // 8192

typedef __attribute__((ext_vector_type(16))) __bf16 v16bf;
typedef __attribute__((ext_vector_type(8)))  __bf16 v8bf;
typedef __attribute__((ext_vector_type(8)))  float  v8f;
typedef __attribute__((ext_vector_type(4)))  int    v4i;

#if defined(__gfx1250__) && __has_builtin(__builtin_amdgcn_global_load_async_to_lds_b128)
#define HAS_ASYNC_LDS 1
#else
#define HAS_ASYNC_LDS 0
#endif

#if HAS_ASYNC_LDS
typedef __attribute__((address_space(1))) v4i gas_v4i;
typedef __attribute__((address_space(3))) v4i las_v4i;
#endif

// 16-byte global -> LDS copy; async DMA (ASYNCcnt) on gfx1250 when available.
static __device__ __forceinline__ void copy16_to_lds(void* lds, const void* g) {
#if HAS_ASYNC_LDS
    __builtin_amdgcn_global_load_async_to_lds_b128((gas_v4i*)g, (las_v4i*)lds, 0, 0);
#else
    *(uint4*)lds = *(const uint4*)g;
#endif
}

static __device__ __forceinline__ void async_copy_wait() {
#if HAS_ASYNC_LDS
#if __has_builtin(__builtin_amdgcn_s_wait_asynccnt)
    __builtin_amdgcn_s_wait_asynccnt(0);
#else
    asm volatile("s_wait_asynccnt 0" ::: "memory");
#endif
#endif
}

static __device__ __forceinline__ v16bf cat8(v8bf lo, v8bf hi) {
    return __builtin_shufflevector(lo, hi, 0,1,2,3,4,5,6,7,8,9,10,11,12,13,14,15);
}

static __device__ __forceinline__ v8f wmma_bf16(v16bf a, v16bf b, v8f c) {
    return __builtin_amdgcn_wmma_f32_16x16x32_bf16(false, a, false, b, (short)0, c, false, false);
}

// ---------------- fp32 -> bf16 conversion ----------------
__global__ __launch_bounds__(256) void cvt_f32_bf16(const float* __restrict__ in,
                                                    __bf16* __restrict__ out, int n) {
    int i = blockIdx.x * blockDim.x + threadIdx.x;
    int stride = gridDim.x * blockDim.x;
    for (; i < n; i += stride) out[i] = (__bf16)in[i];
}

// ---------------- bf16 GEMM:  C(MxN) = A(MxK) @ B(KxN) + bias ----------------
// Block tile 128 x BN, BK=32, double-buffered LDS.
// 8 waves: 4 along M (32 rows) x 2 along N (BN/2 cols).
template <int OUT_F32, int BN>
__global__ __launch_bounds__(256) void gemm_bf16(const __bf16* __restrict__ A,
                                                 const __bf16* __restrict__ B,
                                                 const float*  __restrict__ bias,
                                                 void* __restrict__ Cout,
                                                 int M, int N, int K) {
    constexpr int BM = 128, BK = 32;
    constexpr int TC = BN / 32;          // 16-col tiles per wave
    constexpr int NPT = BN / 8;          // B columns staged per thread
    __shared__ __bf16 As[2][BM][BK];     // row-major (m, k)
    __shared__ __bf16 Bs[2][BN][BK];     // transposed (n, k)

    const int tid  = threadIdx.x;
    const int lane = tid & 31;
    const int wid  = tid >> 5;           // 0..7
    const int wm   = wid & 3;            // wave row group (32 rows)
    const int wn   = wid >> 2;           // wave col group (BN/2 cols)
    const int m0   = blockIdx.y * BM;
    const int n0   = blockIdx.x * BN;
    const int lg   = lane >> 4;          // half-wave group 0/1
    const int lm   = lane & 15;

    v8f acc[2][TC];
    #pragma unroll
    for (int i = 0; i < 2; ++i)
        #pragma unroll
        for (int j = 0; j < TC; ++j)
            acc[i][j] = (v8f){0.f,0.f,0.f,0.f,0.f,0.f,0.f,0.f};

    const int arow  = tid >> 1;          // A staging: 2 threads per row
    const int ahalf = tid & 1;
    const int bk    = tid & 31;          // B staging coords
    const int bng   = tid >> 5;

    auto stage = [&](int buf, int k0) {
        // A tile (128 x 32 bf16) via async LDS DMA: 32 B per thread.
        const __bf16* srcA = A + (size_t)(m0 + arow) * K + k0 + ahalf * 16;
        copy16_to_lds(&As[buf][arow][ahalf * 16], srcA);
        copy16_to_lds(&As[buf][arow][ahalf * 16 + 8], srcA + 8);
        __builtin_prefetch(srcA + BK, 0, 1);
        // B tile transposed: thread owns one k, NPT consecutive n.
        const __bf16* srcB = B + (size_t)(k0 + bk) * N + n0 + bng * NPT;
        #pragma unroll
        for (int j = 0; j < NPT; ++j) Bs[buf][bng * NPT + j][bk] = srcB[j];
    };

    const int nk = K / BK;
    stage(0, 0);

    for (int i = 0; i < nk; ++i) {
        const int p = i & 1;
        async_copy_wait();
        __syncthreads();
        if (i + 1 < nk) stage(p ^ 1, (i + 1) * BK);   // DMA next tile during compute

        v16bf afrag[2], bfrag[TC];
        #pragma unroll
        for (int tr = 0; tr < 2; ++tr) {
            const __bf16* ptr = &As[p][wm * 32 + tr * 16 + lm][lg * 8];
            afrag[tr] = cat8(*(const v8bf*)ptr, *(const v8bf*)(ptr + 16));
        }
        #pragma unroll
        for (int tc = 0; tc < TC; ++tc) {
            const __bf16* ptr = &Bs[p][wn * (BN / 2) + tc * 16 + lm][lg * 8];
            bfrag[tc] = cat8(*(const v8bf*)ptr, *(const v8bf*)(ptr + 16));
        }
        #pragma unroll
        for (int tr = 0; tr < 2; ++tr)
            #pragma unroll
            for (int tc = 0; tc < TC; ++tc)
                acc[tr][tc] = wmma_bf16(afrag[tr], bfrag[tc], acc[tr][tc]);
    }

    // Epilogue: C layout — lane l, vgpr r -> row = 8*(l/16)+r, col = l%16.
    #pragma unroll
    for (int tr = 0; tr < 2; ++tr)
        #pragma unroll
        for (int tc = 0; tc < TC; ++tc)
            #pragma unroll
            for (int r = 0; r < 8; ++r) {
                int grow = m0 + wm * 32 + tr * 16 + lg * 8 + r;
                int gcol = n0 + wn * (BN / 2) + tc * 16 + lm;
                float v = acc[tr][tc][r] + bias[gcol];
                if (OUT_F32) ((float*)Cout)[(size_t)grow * N + gcol] = v;
                else        ((__bf16*)Cout)[(size_t)grow * N + gcol] = (__bf16)v;
            }
}

// ---------------- Flash attention (single KV head, non-causal) ----------------
// Grid: (SEQ/128) * NHQ * BATCH blocks. 8 waves; each wave owns 16 query rows.
// K/V staging double-buffered: async DMA overlaps the WMMA/softmax pipeline.
__global__ __launch_bounds__(256) void mqa_attn(const __bf16* __restrict__ Q,   // (ROWS, 1024)
                                                const __bf16* __restrict__ Km,  // (ROWS, 64)
                                                const __bf16* __restrict__ Vm,  // (ROWS, 64)
                                                __bf16* __restrict__ O) {       // (ROWS, 1024)
    constexpr int BQ = 128, BKEY = 64;
    __shared__ __bf16 Ks[2][BKEY][DH];     // key-major      16 KB
    __shared__ __bf16 Vt[2][DH][BKEY];     // dim-major      16 KB
    __shared__ __bf16 Ps[8][16][BKEY];     // per-wave P     16 KB

    const int tid  = threadIdx.x;
    const int lane = tid & 31;
    const int wid  = tid >> 5;
    const int lg   = lane >> 4;
    const int lm   = lane & 15;

    const int blk = blockIdx.x;
    const int qb  = blk % (SEQ / BQ);
    const int h   = (blk / (SEQ / BQ)) % NHQ;
    const int b   = blk / ((SEQ / BQ) * NHQ);
    const int qrow0 = qb * BQ + wid * 16;

    // Q fragments (16x64 tile -> two 16x32 A fragments), loaded once.
    v16bf qf[2];
    {
        const __bf16* qbase = Q + (size_t)(b * SEQ + qrow0 + lm) * D_MODEL + h * DH;
        #pragma unroll
        for (int d = 0; d < 2; ++d) {
            const __bf16* ptr = qbase + d * 32 + lg * 8;
            qf[d] = cat8(*(const v8bf*)ptr, *(const v8bf*)(ptr + 16));
        }
    }

    float m_i[8], l_i[8];
    #pragma unroll
    for (int r = 0; r < 8; ++r) { m_i[r] = -1e30f; l_i[r] = 0.f; }
    v8f o_acc[4];
    #pragma unroll
    for (int n = 0; n < 4; ++n) o_acc[n] = (v8f){0.f,0.f,0.f,0.f,0.f,0.f,0.f,0.f};

    const float scale = 0.125f;  // 1/sqrt(64)

    auto stageKV = [&](int buf, int kb0) {
        // K block row-major via async LDS DMA: 4 threads per key, 32 B each.
        int key = tid >> 2, part = tid & 3;
        const __bf16* srcK = Km + (size_t)(b * SEQ + kb0 + key) * DH + part * 16;
        copy16_to_lds(&Ks[buf][key][part * 16], srcK);
        copy16_to_lds(&Ks[buf][key][part * 16 + 8], srcK + 8);
        // V transposed: thread owns one key, 16 dims.
        int vkey = tid & 63, d0 = (tid >> 6) * 16;
        const __bf16* srcV = Vm + (size_t)(b * SEQ + kb0 + vkey) * DH + d0;
        #pragma unroll
        for (int j = 0; j < 16; ++j) Vt[buf][d0 + j][vkey] = srcV[j];
    };

    const int nkb = SEQ / BKEY;
    stageKV(0, 0);

    for (int i = 0; i < nkb; ++i) {
        const int p = i & 1;
        async_copy_wait();
        __syncthreads();
        if (i + 1 < nkb) stageKV(p ^ 1, (i + 1) * BKEY);   // DMA next block during compute

        // S = Q @ K^T for 64 keys: 4 col tiles x 2 d-chunks.
        v8f s[4];
        #pragma unroll
        for (int t = 0; t < 4; ++t) {
            s[t] = (v8f){0.f,0.f,0.f,0.f,0.f,0.f,0.f,0.f};
            #pragma unroll
            for (int d = 0; d < 2; ++d) {
                const __bf16* ptr = &Ks[p][t * 16 + lm][d * 32 + lg * 8];
                v16bf bf = cat8(*(const v8bf*)ptr, *(const v8bf*)(ptr + 16));
                s[t] = wmma_bf16(qf[d], bf, s[t]);
            }
        }

        // Online softmax (per row; rows live in vgpr index r, uniform across 16-lane group).
        #pragma unroll
        for (int r = 0; r < 8; ++r) {
            float a0 = s[0][r] * scale, a1 = s[1][r] * scale;
            float a2 = s[2][r] * scale, a3 = s[3][r] * scale;
            s[0][r] = a0; s[1][r] = a1; s[2][r] = a2; s[3][r] = a3;
            float mx = fmaxf(fmaxf(a0, a1), fmaxf(a2, a3));
            #pragma unroll
            for (int m = 1; m <= 8; m <<= 1)
                mx = fmaxf(mx, __shfl_xor(mx, m, 32));
            float mnew  = fmaxf(m_i[r], mx);
            float alpha = __expf(m_i[r] - mnew);
            m_i[r] = mnew;
            float rs = 0.f;
            #pragma unroll
            for (int t = 0; t < 4; ++t) {
                float pv = __expf(s[t][r] - mnew);
                s[t][r] = pv;
                rs += pv;
            }
            #pragma unroll
            for (int m = 1; m <= 8; m <<= 1)
                rs += __shfl_xor(rs, m, 32);
            l_i[r] = l_i[r] * alpha + rs;
            #pragma unroll
            for (int n = 0; n < 4; ++n) o_acc[n][r] *= alpha;
        }

        // Spill P (bf16) to per-wave LDS to re-enter A-matrix layout.
        #pragma unroll
        for (int t = 0; t < 4; ++t)
            #pragma unroll
            for (int r = 0; r < 8; ++r)
                Ps[wid][lg * 8 + r][t * 16 + lm] = (__bf16)s[t][r];

        // O += P @ V : 2 key chunks x 4 output dim tiles.
        #pragma unroll
        for (int kc = 0; kc < 2; ++kc) {
            const __bf16* pp = &Ps[wid][lm][kc * 32 + lg * 8];
            v16bf af = cat8(*(const v8bf*)pp, *(const v8bf*)(pp + 16));
            #pragma unroll
            for (int n = 0; n < 4; ++n) {
                const __bf16* vp = &Vt[p][n * 16 + lm][kc * 32 + lg * 8];
                v16bf bf = cat8(*(const v8bf*)vp, *(const v8bf*)(vp + 16));
                o_acc[n] = wmma_bf16(af, bf, o_acc[n]);
            }
        }
    }

    // Normalize and store (ROWS, h*64 + dim) as bf16.
    #pragma unroll
    for (int n = 0; n < 4; ++n)
        #pragma unroll
        for (int r = 0; r < 8; ++r) {
            int row = qrow0 + lg * 8 + r;
            float v = o_acc[n][r] / l_i[r];
            O[(size_t)(b * SEQ + row) * D_MODEL + h * DH + n * 16 + lm] = (__bf16)v;
        }
}

// ---------------- launch ----------------
extern "C" void kernel_launch(void* const* d_in, const int* in_sizes, int n_in,
                              void* d_out, int out_size, void* d_ws, size_t ws_size,
                              hipStream_t stream) {
    const float* x   = (const float*)d_in[0];
    const float* W_q = (const float*)d_in[1];
    const float* b_q = (const float*)d_in[2];
    const float* W_k = (const float*)d_in[3];
    const float* b_k = (const float*)d_in[4];
    const float* W_v = (const float*)d_in[5];
    const float* b_v = (const float*)d_in[6];
    const float* W_o = (const float*)d_in[7];
    const float* b_o = (const float*)d_in[8];

    char* ws = (char*)d_ws;
    size_t off = 0;
    auto alloc = [&](size_t bytes) { char* p = ws + off; off += (bytes + 255) & ~(size_t)255; return p; };
    __bf16* xb  = (__bf16*)alloc((size_t)ROWS * D_MODEL * 2);
    __bf16* Wqb = (__bf16*)alloc((size_t)D_MODEL * D_MODEL * 2);
    __bf16* Wkb = (__bf16*)alloc((size_t)D_MODEL * DH * 2);
    __bf16* Wvb = (__bf16*)alloc((size_t)D_MODEL * DH * 2);
    __bf16* Wob = (__bf16*)alloc((size_t)D_MODEL * D_MODEL * 2);
    __bf16* qb  = (__bf16*)alloc((size_t)ROWS * D_MODEL * 2);
    __bf16* kb  = (__bf16*)alloc((size_t)ROWS * DH * 2);
    __bf16* vb  = (__bf16*)alloc((size_t)ROWS * DH * 2);
    __bf16* aob = (__bf16*)alloc((size_t)ROWS * D_MODEL * 2);

    cvt_f32_bf16<<<2048, 256, 0, stream>>>(x,   xb,  ROWS * D_MODEL);
    cvt_f32_bf16<<<512,  256, 0, stream>>>(W_q, Wqb, D_MODEL * D_MODEL);
    cvt_f32_bf16<<<64,   256, 0, stream>>>(W_k, Wkb, D_MODEL * DH);
    cvt_f32_bf16<<<64,   256, 0, stream>>>(W_v, Wvb, D_MODEL * DH);
    cvt_f32_bf16<<<512,  256, 0, stream>>>(W_o, Wob, D_MODEL * D_MODEL);

    // Q/K/V projections (bf16 out, bias fused).
    gemm_bf16<0, 128><<<dim3(D_MODEL / 128, ROWS / 128), 256, 0, stream>>>(xb, Wqb, b_q, qb, ROWS, D_MODEL, D_MODEL);
    gemm_bf16<0, 64><<<dim3(1,              ROWS / 128), 256, 0, stream>>>(xb, Wkb, b_k, kb, ROWS, DH,      D_MODEL);
    gemm_bf16<0, 64><<<dim3(1,              ROWS / 128), 256, 0, stream>>>(xb, Wvb, b_v, vb, ROWS, DH,      D_MODEL);

    // Attention: 16 query blocks * 16 heads * 4 batches.
    mqa_attn<<<(SEQ / 128) * NHQ * BATCH, 256, 0, stream>>>(qb, kb, vb, aob);

    // Output projection (fp32 out with bias).
    gemm_bf16<1, 128><<<dim3(D_MODEL / 128, ROWS / 128), 256, 0, stream>>>(aob, Wob, b_o, (float*)d_out, ROWS, D_MODEL, D_MODEL);
}
